// SceneGenerator_88132728913975
// MI455X (gfx1250) — matrix operand order repaired and checked by reading, not compiled, not run on hardware
//
#include <hip/hip_runtime.h>

typedef __attribute__((ext_vector_type(16))) _Float16 v16h;
typedef __attribute__((ext_vector_type(8)))  _Float16 v8h;
typedef __attribute__((ext_vector_type(8)))  float    v8f;
typedef __attribute__((ext_vector_type(4)))  int      v4i;

#define NPTS      262144     // B*N = 4*65536
#define HID       128
#define PPB       128        // points per block
#define THREADS   256

#if __has_builtin(__builtin_amdgcn_global_load_async_to_lds_b128) && \
    __has_builtin(__builtin_amdgcn_s_wait_asynccnt)
#define HAS_ASYNC_LDS 1
#else
#define HAS_ASYNC_LDS 0
#endif

// Address-space-qualified v4i pointers for the async-to-LDS builtin.
typedef __attribute__((address_space(1))) v4i gas_v4i;
typedef __attribute__((address_space(3))) v4i las_v4i;
#define GPTR(p) ((gas_v4i*)(v4i*)(void*)(p))
#define LPTR(p) ((las_v4i*)(v4i*)(void*)(p))

// ---------------------------------------------------------------------------
// Prep: pack W0..W3 (f32, row-major KxN) into f16 B-fragments for
// v_wmma_f32_16x16x32_f16.  B is 32x16 (KxN); this code packs it in contiguous k order:
//   lanes 0-15 : VGPR v holds K=2v,2v+1   (N = lane)
//   lanes16-31 : VGPR v holds K=16+2v,..  (N = lane-16)
// Packed address: [L*16384 + (nt*4+k0)*512 + lane*16 + j] halves.
// W0 row 127 is zero padding (IN_DIM=127 -> K=128).
// ---------------------------------------------------------------------------
__global__ void pack_weights_f16(const float* __restrict__ W0,
                                 const float* __restrict__ W1,
                                 const float* __restrict__ W2,
                                 const float* __restrict__ W3,
                                 _Float16* __restrict__ out) {
    int tid  = blockIdx.x * blockDim.x + threadIdx.x;   // 65536 threads total
    int L    = tid >> 14;
    int rem  = tid & 16383;
    int frag = rem >> 9;            // nt*4 + k0
    int lane = (rem >> 4) & 31;
    int j    = rem & 15;
    int nt   = frag >> 2;
    int k0   = frag & 3;
    int v    = j >> 1;
    int par  = j & 1;
    int krow, ncol;
    if (lane < 16) { krow = k0 * 32 +      2 * v + par; ncol = nt * 16 + lane; }
    else           { krow = k0 * 32 + 16 + 2 * v + par; ncol = nt * 16 + (lane - 16); }
    const float* W = (L == 0) ? W0 : (L == 1) ? W1 : (L == 2) ? W2 : W3;
    float val = (L == 0 && krow == 127) ? 0.0f : W[krow * 128 + ncol];
    out[tid] = (_Float16)val;
}

// ---------------------------------------------------------------------------
// Weight staging: 32KB layer -> LDS.  Async DMA (CDNA5 GLOBAL_LOAD_ASYNC_TO_
// LDS_B128, ASYNCcnt) when the toolchain exposes it; sync copy otherwise.
// Each thread moves 128 contiguous bytes (8 x b128).
// ---------------------------------------------------------------------------
__device__ __forceinline__ void stage_issue(const _Float16* __restrict__ src,
                                            _Float16* dst, int tid) {
    const int base = tid * 64;                // 64 halves = 128B per thread
#if HAS_ASYNC_LDS
    #pragma unroll
    for (int j = 0; j < 8; ++j) {
        __builtin_amdgcn_global_load_async_to_lds_b128(
            GPTR(src + base + j * 8), LPTR(dst + base + j * 8), 0, 0);
    }
#else
    const v8h* s = (const v8h*)(src + base);
    v8h*       d = (v8h*)(dst + base);
    #pragma unroll
    for (int j = 0; j < 8; ++j) d[j] = s[j];
#endif
}

__device__ __forceinline__ void stage_wait() {
#if HAS_ASYNC_LDS
    __builtin_amdgcn_s_wait_asynccnt(0);      // my async LDS writes landed
#endif
    __syncthreads();                          // everyone's landed / buffers free
}

// ---------------------------------------------------------------------------
// Main kernel: 128 points / block, 8 waves, each wave owns 16 points (M-tile).
// Double-buffered weights: layer L+1 DMA overlaps layer L WMMAs.
// ---------------------------------------------------------------------------
__global__ void __launch_bounds__(THREADS)
scene_mlp_wmma(const float* __restrict__ xyz,
               const float* __restrict__ lat,
               const _Float16* __restrict__ wpack,
               const float* __restrict__ b0, const float* __restrict__ b1,
               const float* __restrict__ b2, const float* __restrict__ b3,
               const float* __restrict__ Wa, const float* __restrict__ ba,
               float* __restrict__ out) {
    __shared__ __align__(32) _Float16 hbuf[PPB * HID];     // 32KB activations
    __shared__ __align__(32) _Float16 wbuf[2][HID * HID];  // 2x32KB weights
    __shared__ float bbuf[4 * HID];                        // biases
    __shared__ float red[THREADS];                         // head reduction

    const int tid  = threadIdx.x;
    const int lane = tid & 31;
    const int wv   = tid >> 5;
    const int blk0 = blockIdx.x * PPB;

    // Kick off layer-0 weight DMA immediately; it overlaps all of phase 1.
    stage_issue(wpack, &wbuf[0][0], tid);

    // ---------------- Phase 1a: bilinear grid sample -> dims 0..63 ----------
    {
        int   lp    = tid >> 1;
        int   cbase = (tid & 1) * 32;
        int   p     = blk0 + lp;
        int   b     = p >> 16;                 // p / 65536
        float X = xyz[p * 3 + 0];
        float Z = xyz[p * 3 + 2];
        float px = (X + 1.0f) * 8.0f - 0.5f;   // (g+1)*W/2 - 0.5, W=16
        float py = (Z + 1.0f) * 8.0f - 0.5f;
        float x0f = floorf(px), y0f = floorf(py);
        float fx = px - x0f,    fy = py - y0f;
        int x0 = (int)x0f, y0 = (int)y0f;
        int x1 = x0 + 1,   y1 = y0 + 1;
        int x0c = min(max(x0, 0), 15), x1c = min(max(x1, 0), 15);
        int y0c = min(max(y0, 0), 15), y1c = min(max(y1, 0), 15);
        float vx0 = (x0 == x0c) ? 1.0f : 0.0f, vx1 = (x1 == x1c) ? 1.0f : 0.0f;
        float vy0 = (y0 == y0c) ? 1.0f : 0.0f, vy1 = (y1 == y1c) ? 1.0f : 0.0f;
        float w00 = (1.0f - fx) * (1.0f - fy) * vx0 * vy0;
        float w01 = (1.0f - fx) * fy          * vx0 * vy1;
        float w10 = fx * (1.0f - fy)          * vx1 * vy0;
        float w11 = fx * fy                   * vx1 * vy1;
        const float* img = lat + (size_t)b * 64 * 256;
        int o00 = y0c * 16 + x0c, o01 = y1c * 16 + x0c;
        int o10 = y0c * 16 + x1c, o11 = y1c * 16 + x1c;
        #pragma unroll 4
        for (int c = cbase; c < cbase + 32; ++c) {
            const float* pc = img + c * 256;
            float vz = w00 * pc[o00] + w01 * pc[o01] + w10 * pc[o10] + w11 * pc[o11];
            hbuf[lp * HID + c] = (_Float16)vz;
        }
    }

    // ---------------- Phase 1b: positional encoding -> dims 64..127 ---------
    if (tid < PPB) {
        int   lp = tid, p = blk0 + lp;
        float X = xyz[p * 3 + 0], Y = xyz[p * 3 + 1], Z = xyz[p * 3 + 2];
        float tx = (X + 1.0f) * 8.0f;          // (x+1)*0.5*16
        float tz = (Z + 1.0f) * 8.0f;
        float lcx = (tx - rintf(tx - 0.5f)) * 2.0f - 1.0f;
        float lcz = (tz - rintf(tz - 0.5f)) * 2.0f - 1.0f;
        _Float16* hrow = &hbuf[lp * HID];
        hrow[64] = (_Float16)lcx;
        hrow[65] = (_Float16)Y;
        hrow[66] = (_Float16)lcz;
        float c3[3] = {lcx, Y, lcz};
        #pragma unroll
        for (int f = 0; f < 10; ++f) {
            float fr = (float)(1 << f);
            #pragma unroll
            for (int d = 0; d < 3; ++d) {
                float ang = c3[d] * fr;
                hrow[67 + f * 3 + d] = (_Float16)__sinf(ang);
                hrow[97 + f * 3 + d] = (_Float16)__cosf(ang);
            }
        }
        hrow[127] = (_Float16)0.0f;            // K padding for layer 0
    }

    // biases -> LDS
    for (int i = tid; i < 4 * HID; i += THREADS) {
        const float* bp = (i < 128) ? b0 : (i < 256) ? b1 : (i < 384) ? b2 : b3;
        bbuf[i] = bp[i & 127];
    }

    // ---------------- MLP layers: h = relu(h @ W + b), all via WMMA ---------
    const int m    = wv * 16 + (lane & 15);    // A-fragment row for this lane
    const int aoff = (lane >> 4) * 16;         // K sub-offset per half-wave

    for (int L = 0; L < 4; ++L) {
        // Wait: async weights for layer L landed; all waves done with the
        // other buffer (consumed in layer L-1); hbuf ready (L==0).
        stage_wait();

        // Prefetch next layer's weights into the free buffer; the DMA (or
        // copy) overlaps this layer's WMMA chain.
        if (L < 3) stage_issue(wpack + (L + 1) * 16384, &wbuf[(L + 1) & 1][0], tid);

        const _Float16* wl = &wbuf[L & 1][0];

        // Load all four 16x32 A fragments for this wave's 16 points.
        v16h a[4];
        #pragma unroll
        for (int k = 0; k < 4; ++k) {
            union { v16h v; v8h h[2]; } u;
            u.h[0] = *(const v8h*)&hbuf[m * HID + k * 32 + aoff];
            u.h[1] = *(const v8h*)&hbuf[m * HID + k * 32 + aoff + 8];
            a[k] = u.v;
        }

        const int rowb = wv * 16 + ((lane < 16) ? 0 : 8);  // C/D row base
        const int col  = (lane & 15);
        #pragma unroll
        for (int nt = 0; nt < 8; ++nt) {
            float bias = bbuf[L * HID + nt * 16 + col];
            v8f acc = {bias, bias, bias, bias, bias, bias, bias, bias};
            #pragma unroll
            for (int k = 0; k < 4; ++k) {
                v16h bf = *(const v16h*)&wl[(nt * 4 + k) * 512 + lane * 16];
                acc = __builtin_amdgcn_wmma_f32_16x16x32_f16(
                          false, a[k], false, bf, (short)0, acc, false, false);
            }
            #pragma unroll
            for (int r = 0; r < 8; ++r) {                  // ReLU + writeback
                float v = acc[r];
                v = v > 0.0f ? v : 0.0f;
                hbuf[(rowb + r) * HID + nt * 16 + col] = (_Float16)v;
            }
        }
    }

    // ---------------- Head: alpha = h @ Wa + ba -----------------------------
    {
        int pt = lane & 15, half = lane >> 4;
        const _Float16* hrow = &hbuf[(wv * 16 + pt) * HID + half * 64];
        const float*    wa   = Wa + half * 64;
        float sum = 0.0f;
        #pragma unroll
        for (int d = 0; d < 64; ++d) sum += (float)hrow[d] * wa[d];
        red[tid] = sum;
        __syncthreads();
        if (lane < 16) {
            out[blk0 + wv * 16 + pt] = red[tid] + red[tid + 16] + ba[0];
        }
    }
}

// ---------------------------------------------------------------------------
extern "C" void kernel_launch(void* const* d_in, const int* in_sizes, int n_in,
                              void* d_out, int out_size, void* d_ws, size_t ws_size,
                              hipStream_t stream) {
    const float* xyz = (const float*)d_in[0];
    const float* lat = (const float*)d_in[1];
    const float* W0  = (const float*)d_in[2];
    const float* b0  = (const float*)d_in[3];
    const float* W1  = (const float*)d_in[4];
    const float* b1  = (const float*)d_in[5];
    const float* W2  = (const float*)d_in[6];
    const float* b2  = (const float*)d_in[7];
    const float* W3  = (const float*)d_in[8];
    const float* b3  = (const float*)d_in[9];
    const float* Wa  = (const float*)d_in[10];
    const float* ba  = (const float*)d_in[11];
    float* out = (float*)d_out;

    _Float16* wpack = (_Float16*)d_ws;   // 4*16384 halves = 128KB

    pack_weights_f16<<<256, 256, 0, stream>>>(W0, W1, W2, W3, wpack);
    scene_mlp_wmma<<<NPTS / PPB, THREADS, 0, stream>>>(
        xyz, lat, wpack, b0, b1, b2, b3, Wa, ba, out);
}
